// SinkhornCausalAttention_25391846654032
// MI455X (gfx1250) — compile-verified, matrix operand order checked by reading, not run on hardware
//
#include <hip/hip_runtime.h>
#include <math.h>

#define BB    4
#define NH    8
#define HHALF 4
#define TT    4096
#define DHD   64
#define BSZ   128
#define NBUCK 32
#define NTOP  2
#define NEGF  (-3.402823466e38f)
#define SROW  388   // S row stride (floats): 388%64=4 -> conflict-free row-parallel access
#define KROW  72    // K2h row stride (halves): 36 words -> conflict-free 16-lane b128 reads
#define VROW  392   // V2T row stride (halves): 196 words -> conflict-free 16-lane b128 reads

typedef __attribute__((ext_vector_type(16))) _Float16 v16h;
typedef __attribute__((ext_vector_type(8)))  _Float16 v8h;
typedef __attribute__((ext_vector_type(8)))  float    v8f;
typedef __attribute__((ext_vector_type(4)))  float    v4f;

__device__ __forceinline__ int rotsrc(int p, bool rot) {
  return rot ? ((p + BSZ - 1) & (TT - 1)) : p;
}

// ---------------------------------------------------------------------------
// Kernel 1a: per-bucket sums of (rotated) q and k:  bq/bk [bh][32][64]
// ---------------------------------------------------------------------------
__global__ void k_bsum(const float* __restrict__ q, const float* __restrict__ k,
                       float* __restrict__ bq, float* __restrict__ bk) {
  const int u  = blockIdx.x;
  const int bh = blockIdx.y;
  const int d  = threadIdx.x;                 // 0..63
  const bool rot = (bh & (NH - 1)) >= HHALF;
  const float* qp = q + (size_t)bh * TT * DHD + d;
  const float* kp = k + (size_t)bh * TT * DHD + d;
  float sq = 0.f, sk = 0.f;
  for (int r = 0; r < BSZ; ++r) {
    const int src = rotsrc(u * BSZ + r, rot);
    sq += qp[(size_t)src * DHD];
    sk += kp[(size_t)src * DHD];
  }
  bq[((size_t)bh * NBUCK + u) * DHD + d] = sq;
  bk[((size_t)bh * NBUCK + u) * DHD + d] = sk;
}

// ---------------------------------------------------------------------------
// Kernel 1b: prefix over buckets + cumavg summaries sq/sk [bh][32][64]
//   sq[u] = (prefix_q + q[u*128]) / (u*128+1)
//   sk[u] = sum_j (prefix_k + running_k_j) / (u*128+j+1)
// ---------------------------------------------------------------------------
__global__ void k_summ(const float* __restrict__ q, const float* __restrict__ k,
                       const float* __restrict__ bq, const float* __restrict__ bk,
                       float* __restrict__ sq, float* __restrict__ sk) {
  const int u  = blockIdx.x;
  const int bh = blockIdx.y;
  const int d  = threadIdx.x;
  const bool rot = (bh & (NH - 1)) >= HHALF;
  float preQ = 0.f, preK = 0.f;
  for (int vb = 0; vb < u; ++vb) {
    preQ += bq[((size_t)bh * NBUCK + vb) * DHD + d];
    preK += bk[((size_t)bh * NBUCK + vb) * DHD + d];
  }
  const float* qp = q + (size_t)bh * TT * DHD + d;
  const float* kp = k + (size_t)bh * TT * DHD + d;
  const int p0 = u * BSZ;
  sq[((size_t)bh * NBUCK + u) * DHD + d] =
      (preQ + qp[(size_t)rotsrc(p0, rot) * DHD]) / (float)(p0 + 1);
  float run = preK, acc = 0.f;
  for (int r = 0; r < BSZ; ++r) {
    run += kp[(size_t)rotsrc(p0 + r, rot) * DHD];
    acc += run / (float)(p0 + r + 1);
  }
  sk[((size_t)bh * NBUCK + u) * DHD + d] = acc;
}

// ---------------------------------------------------------------------------
// Kernel 2: bucket affinity R[32][34] (causal-masked), iterative top-2 softmax
// ---------------------------------------------------------------------------
__global__ void k_topk(const float* __restrict__ sqg, const float* __restrict__ skg,
                       int* __restrict__ idxo, float* __restrict__ wo) {
  __shared__ float sq[NBUCK * DHD];
  __shared__ float sk[NBUCK * DHD];
  const int bh  = blockIdx.x;
  const int tid = threadIdx.x;       // 0..31
  for (int e = tid; e < NBUCK * DHD; e += 32) {
    sq[e] = sqg[(size_t)bh * NBUCK * DHD + e];
    sk[e] = skg[(size_t)bh * NBUCK * DHD + e];
  }
  __syncthreads();
  const int u = tid;
  float R[NBUCK + NTOP];
  for (int j = 0; j < NBUCK + NTOP; ++j) {
    if (j >= u + NTOP) { R[j] = NEGF; continue; }
    if (j < NTOP)      { R[j] = 0.f;  continue; }
    float acc = 0.f;
    for (int d = 0; d < DHD; ++d) acc += sq[u * DHD + d] * sk[(j - NTOP) * DHD + d];
    R[j] = acc * 0.125f;
  }
  const int base = (bh * NBUCK + u) * NTOP;
  for (int pick = 0; pick < NTOP; ++pick) {
    float m = -INFINITY; int bi = 0;
    for (int j = 0; j < NBUCK + NTOP; ++j)
      if (R[j] > m) { m = R[j]; bi = j; }
    float s = 0.f;
    for (int j = 0; j < NBUCK + NTOP; ++j) s += __expf(R[j] - m);
    idxo[base + pick] = bi;
    wo[base + pick]   = 1.0f / s;
    R[bi] = NEGF;
  }
}

// ---------------------------------------------------------------------------
// Kernel 3: bucketed attention. One WG (8 wave32) per (bh, bucket).
// LDS: K2h f16[384][KROW] | V2T f16[64][VROW] | S f32[128][SROW] | stats
// ---------------------------------------------------------------------------
__global__ void __launch_bounds__(256) k_attn(
    const float* __restrict__ q, const float* __restrict__ kg,
    const float* __restrict__ vg, const float* __restrict__ nullk,
    const float* __restrict__ nullv, const int* __restrict__ idxi,
    const float* __restrict__ wi, float* __restrict__ out) {
  extern __shared__ char smem[];
  _Float16* K2h = (_Float16*)smem;                      // 384 * KROW halves
  _Float16* V2T = K2h + 3 * BSZ * KROW;                 // 64  * VROW halves
  float* S    = (float*)(V2T + DHD * VROW);             // 128 * SROW floats
  float* rmax = S + BSZ * SROW;                         // 128
  float* rinv = rmax + BSZ;                             // 128
  float* pm   = rinv + BSZ;                             // 256 (row*2 + half)
  float* ps   = pm + 2 * BSZ;                           // 256

  const int u   = blockIdx.x;
  const int bh  = blockIdx.y;
  const int h   = bh & (NH - 1);
  const bool rot     = (h >= HHALF);
  const bool special = rot && (u == NBUCK - 1);
  const int tid = threadIdx.x;
  const size_t baseT = (size_t)bh * TT * DHD;

  const int   id0 = idxi[(bh * NBUCK + u) * 2 + 0];
  const int   id1 = idxi[(bh * NBUCK + u) * 2 + 1];
  const float w0  = wi[(bh * NBUCK + u) * 2 + 0];
  const float w1  = wi[(bh * NBUCK + u) * 2 + 1];

  // ---- Phase 1: stage K2 (row-major, padded) and V2 (transposed) as f16 ---
  for (int e = tid; e < 3 * BSZ * DHD; e += 256) {
    const int row = e >> 6, d = e & 63;
    float kv, vv;
    if (row < 2 * BSZ) {                                  // gathered picks
      const int   pick = row >> 7;
      const int   r    = row & (BSZ - 1);
      const int   id   = pick ? id1 : id0;
      const float wt   = pick ? w1 : w0;
      if (id < NTOP) {                                    // null bucket
        kv = nullk[h * DHD + d] * wt;
        vv = nullv[h * DHD + d] * wt;
      } else {
        const int src = rotsrc((id - NTOP) * BSZ + r, rot);
        kv = kg[baseT + (size_t)src * DHD + d] * wt;
        vv = vg[baseT + (size_t)src * DHD + d] * wt;
      }
    } else {                                              // own bucket
      const int src = rotsrc(u * BSZ + (row - 2 * BSZ), rot);
      kv = kg[baseT + (size_t)src * DHD + d];
      vv = vg[baseT + (size_t)src * DHD + d];
    }
    K2h[row * KROW + d]  = (_Float16)kv;
    V2T[d * VROW + row]  = (_Float16)vv;
  }
  __syncthreads();

  const int wave = tid >> 5;       // wave32
  const int lane = tid & 31;
  const int m0   = wave * 16;      // 16-row band per wave
  const int lm   = lane & 15;
  const int koff = (lane >> 4) * 8;
  const int rhalf = (lane >> 4) * 8;

  // A fragments of Q straight from global (this lane's row is fixed)
  v16h aQ[2];
  {
    const int src = rotsrc(u * BSZ + m0 + lm, rot);
    const float* qp = q + baseT + (size_t)src * DHD;
#pragma unroll
    for (int c = 0; c < 2; ++c) {
      const int c0 = c * 32 + koff;
      v4f x0 = *(const v4f*)(qp + c0);
      v4f x1 = *(const v4f*)(qp + c0 + 4);
      v4f y0 = *(const v4f*)(qp + c0 + 16);
      v4f y1 = *(const v4f*)(qp + c0 + 20);
#pragma unroll
      for (int e = 0; e < 4; ++e) {
        aQ[c][e]      = (_Float16)x0[e];
        aQ[c][4 + e]  = (_Float16)x1[e];
        aQ[c][8 + e]  = (_Float16)y0[e];
        aQ[c][12 + e] = (_Float16)y1[e];
      }
    }
  }

  // ---- Phase 2: S = mask(Q K2^T * DH^-0.5), branchless mask --------------
  const int jcol = lm;  // column within 16-wide tile
  for (int nb = 0; nb < 24; ++nb) {
    v8f acc = {};
#pragma unroll
    for (int c = 0; c < 2; ++c) {
      const _Float16* p0 = &K2h[(nb * 16 + lm) * KROW + c * 32 + koff];
      v8h lo = *(const v8h*)p0;
      v8h hi = *(const v8h*)(p0 + 16);
      v16h bK;
#pragma unroll
      for (int e = 0; e < 8; ++e) { bK[e] = lo[e]; bK[8 + e] = hi[e]; }
      acc = __builtin_amdgcn_wmma_f32_16x16x32_f16(false, aQ[c], false, bK,
                                                   (short)0, acc, false, false);
    }
    const int j  = nb * 16 + jcol;
    const int jm = j - 2 * BSZ;
    const int jgt = (j > 2 * BSZ) ? 1 : 0;      // per-lane, branch-free
    const int nsp = special ? 0 : 1;            // uniform
#pragma unroll
    for (int r = 0; r < 8; ++r) {
      const int gi = m0 + rhalf + r;
      // base: jm <= gi ; special removes rows gi>0 for j <= 256
      const int ok = (jm <= gi) & (jgt | (gi == 0) | nsp);
      S[gi * SROW + j] = ok ? (acc[r] * 0.125f) : NEGF;
    }
  }

  // ---- Phase 3: row softmax stats, lane pairs split the 384 columns -------
  {
    const int rowi = m0 + lm;
    const int hf   = lane >> 4;
    const float* Sp = &S[rowi * SROW + hf * 192];
    float m = -INFINITY;
    for (int j = 0; j < 192; ++j) m = fmaxf(m, Sp[j]);
    float s = 0.f;
    for (int j = 0; j < 192; ++j) s += __expf(Sp[j] - m);
    pm[rowi * 2 + hf] = m;
    ps[rowi * 2 + hf] = s;
  }
  if (lane < 16) {
    const int rowi = m0 + lane;
    const float ma = pm[rowi * 2 + 0], mb = pm[rowi * 2 + 1];
    const float M  = fmaxf(ma, mb);
    const float s  = ps[rowi * 2 + 0] * __expf(ma - M) +
                     ps[rowi * 2 + 1] * __expf(mb - M);
    rmax[rowi] = M;
    rinv[rowi] = 1.0f / s;
  }
  __syncthreads();

  // ---- Phase 4: O = exp(S - rmax) @ V2, exp fused into A-fragment build ---
  const float rowm = rmax[m0 + lm];   // this lane's P row is fixed
  v8f accO[4] = {{}, {}, {}, {}};
  for (int kb = 0; kb < 12; ++kb) {
    v16h aP;
    const float* Sp = &S[(m0 + lm) * SROW + kb * 32 + koff];
    v4f s0 = *(const v4f*)(Sp);
    v4f s1 = *(const v4f*)(Sp + 4);
    v4f s2 = *(const v4f*)(Sp + 16);
    v4f s3 = *(const v4f*)(Sp + 20);
#pragma unroll
    for (int e = 0; e < 4; ++e) {
      aP[e]      = (_Float16)__expf(s0[e] - rowm);
      aP[4 + e]  = (_Float16)__expf(s1[e] - rowm);
      aP[8 + e]  = (_Float16)__expf(s2[e] - rowm);
      aP[12 + e] = (_Float16)__expf(s3[e] - rowm);
    }
#pragma unroll
    for (int fb = 0; fb < 4; ++fb) {
      const _Float16* p0 = &V2T[(fb * 16 + lm) * VROW + kb * 32 + koff];
      v8h lo = *(const v8h*)p0;
      v8h hi = *(const v8h*)(p0 + 16);
      v16h bV;
#pragma unroll
      for (int e = 0; e < 8; ++e) { bV[e] = lo[e]; bV[8 + e] = hi[e]; }
      accO[fb] = __builtin_amdgcn_wmma_f32_16x16x32_f16(false, aP, false, bV,
                                                        (short)0, accO[fb], false, false);
    }
  }

  // ---- Phase 5: normalize, inverse-rotate, store --------------------------
  float rv[8]; int tf[8];
#pragma unroll
  for (int r = 0; r < 8; ++r) {
    const int gi = m0 + rhalf + r;
    rv[r] = rinv[gi];
    tf[r] = rotsrc(u * BSZ + gi, rot);   // roll back right by BSZ-1
  }
#pragma unroll
  for (int fb = 0; fb < 4; ++fb) {
    const int f = fb * 16 + lm;
#pragma unroll
    for (int r = 0; r < 8; ++r)
      out[baseT + (size_t)tf[r] * DHD + f] = accO[fb][r] * rv[r];
  }
}

// ---------------------------------------------------------------------------
extern "C" void kernel_launch(void* const* d_in, const int* in_sizes, int n_in,
                              void* d_out, int out_size, void* d_ws, size_t ws_size,
                              hipStream_t stream) {
  const float* q     = (const float*)d_in[0];
  const float* k     = (const float*)d_in[1];
  const float* v     = (const float*)d_in[2];
  const float* nullk = (const float*)d_in[3];
  const float* nullv = (const float*)d_in[4];
  float* out = (float*)d_out;

  float* bq   = (float*)d_ws;                 // [32][32][64]
  float* bk   = bq + 32 * 32 * 64;
  float* sq   = bk + 32 * 32 * 64;
  float* sk   = sq + 32 * 32 * 64;
  int*   idxp = (int*)(sk + 32 * 32 * 64);    // [32][32][2]
  float* wp   = (float*)(idxp + 32 * 32 * 2); // [32][32][2]

  k_bsum<<<dim3(32, 32), dim3(64), 0, stream>>>(q, k, bq, bk);
  k_summ<<<dim3(32, 32), dim3(64), 0, stream>>>(q, k, bq, bk, sq, sk);
  k_topk<<<dim3(32), dim3(32), 0, stream>>>(sq, sk, idxp, wp);

  const size_t smem = (size_t)(3 * BSZ * KROW + DHD * VROW) * sizeof(_Float16)
                    + (size_t)(BSZ * SROW + 6 * BSZ) * sizeof(float);
  hipFuncSetAttribute(reinterpret_cast<const void*>(k_attn),
                      hipFuncAttributeMaxDynamicSharedMemorySize, (int)smem);
  k_attn<<<dim3(32, 32), dim3(256), smem, stream>>>(q, k, v, nullk, nullv,
                                                    idxp, wp, out);
}
